// LimLayer_5875515261230
// MI455X (gfx1250) — compile-verified, matrix operand order checked by reading, not run on hardware
//
#include <hip/hip_runtime.h>
#include <hip/hip_bf16.h>

typedef _Float16 v16h __attribute__((ext_vector_type(16)));
typedef _Float16 v8h  __attribute__((ext_vector_type(8)));
typedef float    v8f  __attribute__((ext_vector_type(8)));

#define WMMA_F16(a, b, c) \
  __builtin_amdgcn_wmma_f32_16x16x32_f16(false, (a), false, (b), (short)0, (c), false, false)

// ---------------------------------------------------------------------------
// K0: swizzle a row-major f32 weight W[K][NOUT] into WMMA B-fragment order
// (f16). Fragment (kt,nt): 32 lanes x 16 halfs contiguous.
//   lane: col = lane&15 (N), g = lane>>4
//   e<8  -> k = kt*32 + 8g + e ; e>=8 -> k = kt*32 + 16 + 8g + (e-8)
// ---------------------------------------------------------------------------
__global__ void k_swizzle(const float* __restrict__ W, _Float16* __restrict__ dst,
                          int NOUT, int total) {
  int idx = blockIdx.x * blockDim.x + threadIdx.x;
  if (idx >= total) return;
  int e    = idx & 15;
  int lane = (idx >> 4) & 31;
  int frag = idx >> 9;
  int NT   = NOUT >> 4;
  int nt   = frag % NT;
  int kt   = frag / NT;
  int col  = lane & 15, g = lane >> 4;
  int k = kt * 32 + ((e < 8) ? (g * 8 + e) : (16 + g * 8 + (e - 8)));
  int n = nt * 16 + col;
  dst[idx] = (_Float16)W[k * NOUT + n];
}

// ---------------------------------------------------------------------------
// K1: struct-enc MLP. Per 16-node tile (one wave each):
//   h1 = GELU(x @ W1 + b1)   [16,128]   (8 k-steps x 8 n-tiles WMMA)
//   ang = tanh(h1 @ W2 + b2) [16,64]    (4 x 4 WMMA, h1 staged in wave LDS)
//   cs[node][bundle] = (cos ang, sin ang)
// Dynamic LDS: W1 frags (32768h) + W2 frags (8192h) + 4*2048h staging = 96KB
// ---------------------------------------------------------------------------
__global__ __launch_bounds__(128) void k_structenc(
    const float* __restrict__ x, const float* __restrict__ b1,
    const float* __restrict__ b2, const _Float16* __restrict__ w1h,
    const _Float16* __restrict__ w2h, float* __restrict__ cs, int ntiles) {
  extern __shared__ _Float16 sm[];
  _Float16* sw1 = sm;           // 32768 halfs
  _Float16* sw2 = sm + 32768;   // 8192 halfs
  _Float16* stg = sm + 40960;   // 4 waves * 2048 halfs

  {  // cooperative weight load (b128 copies)
    const uint4* s1 = (const uint4*)w1h;
    uint4* d1 = (uint4*)sw1;
    for (int i = threadIdx.x; i < 32768 / 8; i += blockDim.x) d1[i] = s1[i];
    const uint4* s2 = (const uint4*)w2h;
    uint4* d2 = (uint4*)sw2;
    for (int i = threadIdx.x; i < 8192 / 8; i += blockDim.x) d2[i] = s2[i];
  }
  __syncthreads();

  const int lane = threadIdx.x & 31;
  const int wave = threadIdx.x >> 5;
  const int col = lane & 15, g = lane >> 4;
  _Float16* st = stg + wave * 2048;  // staging layout: [128 K][16 rows]

  for (int tile = blockIdx.x * 4 + wave; tile < ntiles; tile += gridDim.x * 4) {
    const int row = tile * 16 + col;

    // ---- GEMM1: x[16,256] @ W1[256,128] ----
    v8f acc[8];
#pragma unroll
    for (int i = 0; i < 8; ++i) acc[i] = (v8f){};
#pragma unroll
    for (int kt = 0; kt < 8; ++kt) {
      const float* xr = x + (size_t)row * 256 + kt * 32 + g * 8;
      float4 xa = *(const float4*)(xr);
      float4 xb = *(const float4*)(xr + 4);
      float4 xc = *(const float4*)(xr + 16);
      float4 xd = *(const float4*)(xr + 20);
      v16h a;
      a[0] = (_Float16)xa.x; a[1] = (_Float16)xa.y; a[2] = (_Float16)xa.z; a[3] = (_Float16)xa.w;
      a[4] = (_Float16)xb.x; a[5] = (_Float16)xb.y; a[6] = (_Float16)xb.z; a[7] = (_Float16)xb.w;
      a[8] = (_Float16)xc.x; a[9] = (_Float16)xc.y; a[10] = (_Float16)xc.z; a[11] = (_Float16)xc.w;
      a[12] = (_Float16)xd.x; a[13] = (_Float16)xd.y; a[14] = (_Float16)xd.z; a[15] = (_Float16)xd.w;
#pragma unroll
      for (int nt = 0; nt < 8; ++nt) {
        v16h b = *(const v16h*)(sw1 + ((kt * 8 + nt) * 32 + lane) * 16);
        acc[nt] = WMMA_F16(a, b, acc[nt]);
      }
    }

    // ---- bias + exact GELU, stage h1 as f16 (one b128 store per nt) ----
#pragma unroll
    for (int nt = 0; nt < 8; ++nt) {
      int n = nt * 16 + col;
      float bias = b1[n];
      v8h hh;
#pragma unroll
      for (int r = 0; r < 8; ++r) {
        float v = acc[nt][r] + bias;
        v = 0.5f * v * (1.0f + erff(v * 0.70710678118654752f));
        hh[r] = (_Float16)v;
      }
      *(v8h*)(st + n * 16 + g * 8) = hh;  // rows 8g..8g+7 of column n
    }

    // ---- GEMM2: h1[16,128] @ W2[128,64] ----
    v8f acc2[4];
#pragma unroll
    for (int i = 0; i < 4; ++i) acc2[i] = (v8f){};
#pragma unroll
    for (int kt = 0; kt < 4; ++kt) {
      v16h a;
#pragma unroll
      for (int e = 0; e < 8; ++e) {
        int k0 = kt * 32 + g * 8 + e;
        int k1 = kt * 32 + 16 + g * 8 + e;
        a[e]     = st[k0 * 16 + col];
        a[8 + e] = st[k1 * 16 + col];
      }
#pragma unroll
      for (int nt = 0; nt < 4; ++nt) {
        v16h b = *(const v16h*)(sw2 + ((kt * 4 + nt) * 32 + lane) * 16);
        acc2[nt] = WMMA_F16(a, b, acc2[nt]);
      }
    }

    // ---- tanh -> angle -> (cos,sin) per (node,bundle) ----
#pragma unroll
    for (int nt = 0; nt < 4; ++nt) {
      int bidx = nt * 16 + col;
      float bias = b2[bidx];
#pragma unroll
      for (int r = 0; r < 8; ++r) {
        int node = tile * 16 + r + 8 * g;
        float ang = tanhf(acc2[nt][r] + bias);
        float sn, cn;
        sincosf(ang, &sn, &cn);
        float2 p; p.x = cn; p.y = sn;
        *(float2*)(cs + (size_t)node * 128 + bidx * 2) = p;
      }
    }
  }
}

// ---------------------------------------------------------------------------
// K2: rotate fibers of x in-register, big GEMM [16,256]@[256,256] (lin_w in
// LDS, 128 WMMA/tile), add bias, reduce rows in-fragment and accumulate
// per-graph sums (batch is sorted -> fast path is one atomic per column).
// Dynamic LDS: 65536 halfs = 128KB
// ---------------------------------------------------------------------------
__global__ __launch_bounds__(128) void k_main(
    const float* __restrict__ x, const _Float16* __restrict__ wlh,
    const float* __restrict__ linb, const float* __restrict__ cs,
    const int* __restrict__ batch, float* __restrict__ sums, int ntiles) {
  extern __shared__ _Float16 sm[];
  {
    const uint4* s = (const uint4*)wlh;
    uint4* d = (uint4*)sm;
    for (int i = threadIdx.x; i < 65536 / 8; i += blockDim.x) d[i] = s[i];
  }
  __syncthreads();

  const int lane = threadIdx.x & 31;
  const int wave = threadIdx.x >> 5;
  const int col = lane & 15, g = lane >> 4;

  for (int tile = blockIdx.x * 4 + wave; tile < ntiles; tile += gridDim.x * 4) {
    const int row = tile * 16 + col;
    v8f acc[16];
#pragma unroll
    for (int i = 0; i < 16; ++i) acc[i] = (v8f){};

#pragma unroll
    for (int kt = 0; kt < 8; ++kt) {
      v16h a;
#pragma unroll
      for (int h = 0; h < 2; ++h) {
        int chb = kt * 32 + h * 16 + g * 8;  // 8 channels = 2 bundles
        const float* xr = x + (size_t)row * 256 + chb;
        float4 x0 = *(const float4*)xr;
        float4 x1 = *(const float4*)(xr + 4);
        float4 c4 = *(const float4*)(cs + (size_t)row * 128 + (chb >> 2) * 2);
        // bundle0: (c,s)=(c4.x,c4.y); bundle1: (c4.z,c4.w)
        int e0 = h * 8;
        a[e0 + 0] = (_Float16)(c4.x * x0.x - c4.y * x0.z);
        a[e0 + 1] = (_Float16)(c4.x * x0.y - c4.y * x0.w);
        a[e0 + 2] = (_Float16)(c4.y * x0.x + c4.x * x0.z);
        a[e0 + 3] = (_Float16)(c4.y * x0.y + c4.x * x0.w);
        a[e0 + 4] = (_Float16)(c4.z * x1.x - c4.w * x1.z);
        a[e0 + 5] = (_Float16)(c4.z * x1.y - c4.w * x1.w);
        a[e0 + 6] = (_Float16)(c4.w * x1.x + c4.z * x1.z);
        a[e0 + 7] = (_Float16)(c4.w * x1.y + c4.z * x1.w);
      }
#pragma unroll
      for (int nt = 0; nt < 16; ++nt) {
        v16h b = *(const v16h*)(sm + ((kt * 16 + nt) * 32 + lane) * 16);
        acc[nt] = WMMA_F16(a, b, acc[nt]);
      }
    }

    int g0 = batch[tile * 16];
    int g15 = batch[tile * 16 + 15];
    if (g0 == g15) {  // whole tile in one graph: reduce 16 rows in-fragment
#pragma unroll
      for (int nt = 0; nt < 16; ++nt) {
        float bias = linb[nt * 16 + col];
        float p = 0.0f;
#pragma unroll
        for (int r = 0; r < 8; ++r) p += acc[nt][r] + bias;
        p += __shfl_xor(p, 16);  // combine row halves (g=0/1)
        if (lane < 16) atomicAdd(&sums[g0 * 256 + nt * 16 + col], p);
      }
    } else {  // graph boundary inside tile (rare): per-row atomics
#pragma unroll
      for (int nt = 0; nt < 16; ++nt) {
        float bias = linb[nt * 16 + col];
#pragma unroll
        for (int r = 0; r < 8; ++r) {
          int node = tile * 16 + r + 8 * g;
          int gg = batch[node];
          atomicAdd(&sums[gg * 256 + nt * 16 + col], acc[nt][r] + bias);
        }
      }
    }
  }
}

// ---------------------------------------------------------------------------
__global__ void k_count(const int* __restrict__ batch, int* __restrict__ cnt, int n) {
  int i = blockIdx.x * blockDim.x + threadIdx.x;
  if (i < n) atomicAdd(&cnt[batch[i]], 1);
}

__global__ void k_mean(float* __restrict__ sums, const int* __restrict__ cnt) {
  int idx = blockIdx.x * blockDim.x + threadIdx.x;  // 128*256
  int g = idx >> 8;
  int c = cnt[g];
  sums[idx] = sums[idx] / (float)(c > 0 ? c : 1);
}

// ---------------------------------------------------------------------------
// shared helper: h = x + R^T * hmean[batch]  (one float4 = one bundle)
// ---------------------------------------------------------------------------
__device__ __forceinline__ float4 residual_invrot(
    const float* __restrict__ x, const float* __restrict__ hmean,
    const float* __restrict__ cs, int node, int gg, int c4) {
  float4 hm = ((const float4*)(hmean + (size_t)gg * 256))[c4];
  float2 p = *(const float2*)(cs + (size_t)node * 128 + c4 * 2);
  float4 x4 = ((const float4*)(x + (size_t)node * 256))[c4];
  float c = p.x, s = p.y;
  float4 hv;
  hv.x = x4.x + (c * hm.x + s * hm.z);
  hv.y = x4.y + (c * hm.y + s * hm.w);
  hv.z = x4.z + (-s * hm.x + c * hm.z);
  hv.w = x4.w + (-s * hm.y + c * hm.w);
  return hv;
}

// ---------------------------------------------------------------------------
// K4: BatchNorm statistics (sum, sumsq per channel) with register
// accumulation + LDS block reduction; few global atomics per block.
// thread t: channel-quad c4 = t&63, node lane nl = t>>6
// ---------------------------------------------------------------------------
__global__ __launch_bounds__(256) void k_bnstats(
    const float* __restrict__ x, const float* __restrict__ hmean,
    const float* __restrict__ cs, const int* __restrict__ batch,
    float* __restrict__ bnsum, float* __restrict__ bnsq, int n) {
  __shared__ float red[256 * 8];
  int t = threadIdx.x;
  int c4 = t & 63, nl = t >> 6;
  float s0 = 0, s1 = 0, s2 = 0, s3 = 0;
  float q0 = 0, q1 = 0, q2 = 0, q3 = 0;
  for (int node = blockIdx.x * 4 + nl; node < n; node += gridDim.x * 4) {
    int gg = batch[node];
    float4 hv = residual_invrot(x, hmean, cs, node, gg, c4);
    s0 += hv.x; s1 += hv.y; s2 += hv.z; s3 += hv.w;
    q0 += hv.x * hv.x; q1 += hv.y * hv.y; q2 += hv.z * hv.z; q3 += hv.w * hv.w;
  }
  float* rb = red + t * 8;
  rb[0] = s0; rb[1] = s1; rb[2] = s2; rb[3] = s3;
  rb[4] = q0; rb[5] = q1; rb[6] = q2; rb[7] = q3;
  __syncthreads();
  if (t < 64) {
#pragma unroll
    for (int j = 0; j < 8; ++j) {
      float v = red[t * 8 + j] + red[(64 + t) * 8 + j] +
                red[(128 + t) * 8 + j] + red[(192 + t) * 8 + j];
      if (j < 4) atomicAdd(&bnsum[t * 4 + j], v);
      else       atomicAdd(&bnsq[t * 4 + (j - 4)], v);
    }
  }
}

// ---------------------------------------------------------------------------
// K6: per-channel scale/shift from stats, recompute h, normalize, write out.
// ---------------------------------------------------------------------------
__global__ __launch_bounds__(256) void k_finalize(
    const float* __restrict__ x, const float* __restrict__ hmean,
    const float* __restrict__ cs, const int* __restrict__ batch,
    const float* __restrict__ bnsum, const float* __restrict__ bnsq,
    const float* __restrict__ gamma, const float* __restrict__ beta,
    float* __restrict__ out, int n) {
  __shared__ float sa[256], sb[256];
  int t = threadIdx.x;
  {
    float inv_n = 1.0f / (float)n;
    float mu = bnsum[t] * inv_n;
    float var = bnsq[t] * inv_n - mu * mu;
    float iv = rsqrtf(var + 1e-5f);
    float av = gamma[t] * iv;
    sa[t] = av;
    sb[t] = beta[t] - mu * av;
  }
  __syncthreads();
  int c4 = t & 63, nl = t >> 6;
  for (int node = blockIdx.x * 4 + nl; node < n; node += gridDim.x * 4) {
    int gg = batch[node];
    float4 hv = residual_invrot(x, hmean, cs, node, gg, c4);
    float4 o;
    o.x = hv.x * sa[c4 * 4 + 0] + sb[c4 * 4 + 0];
    o.y = hv.y * sa[c4 * 4 + 1] + sb[c4 * 4 + 1];
    o.z = hv.z * sa[c4 * 4 + 2] + sb[c4 * 4 + 2];
    o.w = hv.w * sa[c4 * 4 + 3] + sb[c4 * 4 + 3];
    ((float4*)(out + (size_t)node * 256))[c4] = o;
  }
}

// ---------------------------------------------------------------------------
extern "C" void kernel_launch(void* const* d_in, const int* in_sizes, int n_in,
                              void* d_out, int out_size, void* d_ws, size_t ws_size,
                              hipStream_t stream) {
  const float* x     = (const float*)d_in[0];
  // d_in[1] = edge_index (unused)
  const int*   batch = (const int*)d_in[2];
  const float* se_w1 = (const float*)d_in[3];
  const float* se_b1 = (const float*)d_in[4];
  const float* se_w2 = (const float*)d_in[5];
  const float* se_b2 = (const float*)d_in[6];
  const float* lin_w = (const float*)d_in[7];
  const float* lin_b = (const float*)d_in[8];
  const float* gamma = (const float*)d_in[9];
  const float* beta  = (const float*)d_in[10];

  const int n = in_sizes[0] / 256;   // 100000
  const int ntiles = n / 16;         // 6250 (exact)

  // workspace layout
  char* ws = (char*)d_ws;
  _Float16* w1h = (_Float16*)(ws);             //  64 KB
  _Float16* w2h = (_Float16*)(ws + 65536);     //  16 KB
  _Float16* wlh = (_Float16*)(ws + 81920);     // 128 KB
  float* sums   = (float*)(ws + 212992);       // 128 KB (becomes hmean)
  int*   cnt    = (int*)(ws + 344064);         // 512 B
  float* bnsum  = (float*)(ws + 344576);       // 1 KB
  float* bnsq   = (float*)(ws + 345600);       // 1 KB
  float* cs     = (float*)(ws + 352256);       // n*128*4 = 51.2 MB

  // zero accumulators (sums, cnt, bnsum, bnsq are contiguous)
  hipMemsetAsync(ws + 212992, 0, 131072 + 512 + 1024 + 1024, stream);

  // weight swizzle to WMMA B-fragment layout (f16)
  k_swizzle<<<(32768 + 255) / 256, 256, 0, stream>>>(se_w1, w1h, 128, 32768);
  k_swizzle<<<(8192 + 255) / 256, 256, 0, stream>>>(se_w2, w2h, 64, 8192);
  k_swizzle<<<(65536 + 255) / 256, 256, 0, stream>>>(lin_w, wlh, 256, 65536);

  // struct-enc MLP -> cos/sin per (node,bundle)
  k_structenc<<<512, 128, 98304, stream>>>(x, se_b1, se_b2, w1h, w2h, cs, ntiles);

  // graph node counts
  k_count<<<(n + 255) / 256, 256, 0, stream>>>(batch, cnt, n);

  // rotate + big GEMM + pooled graph sums
  k_main<<<512, 128, 131072, stream>>>(x, wlh, lin_b, cs, batch, sums, ntiles);

  // sums -> means (in place)
  k_mean<<<128, 256, 0, stream>>>(sums, cnt);

  // batchnorm statistics
  k_bnstats<<<512, 256, 0, stream>>>(x, sums, cs, batch, bnsum, bnsq, n);

  // normalize + write output
  k_finalize<<<1024, 256, 0, stream>>>(x, sums, cs, batch, bnsum, bnsq,
                                       gamma, beta, (float*)d_out, n);
}
